// TwoDimAttention_60017872994777
// MI455X (gfx1250) — compile-verified, matrix operand order checked by reading, not run on hardware
//
#include <hip/hip_runtime.h>
#include <hip/hip_bf16.h>

#define B_  8
#define T_  512
#define F_  80
#define TF_ (T_ * F_)

typedef __bf16 bf16;
typedef __attribute__((ext_vector_type(16))) __bf16 bf16x16;
typedef __attribute__((ext_vector_type(8)))  float  f32x8;

union Frag {
    bf16x16 v;
    bf16    h[16];
    uint4   q[2];
};

__device__ __forceinline__ f32x8 wmma_bf16(const Frag& a, const Frag& b, f32x8 c) {
    return __builtin_amdgcn_wmma_f32_16x16x32_bf16(false, a.v, false, b.v,
                                                   (short)0, c, false, false);
}

// 16-bit operand layout (ISA 7.12.2): element i of lane L maps to
// K = (i<8 ? i : i+8) + (L<16 ? 0 : 8); row/col = L%16.
// => per lane, elements [0..7] and [8..15] are two contiguous K-runs of 8.
__device__ __forceinline__ uint4 zero4() {
    uint4 z; z.x = 0u; z.y = 0u; z.z = 0u; z.w = 0u; return z;
}

// ---------------------------------------------------------------------------
// Prep: transpose f32 weights [K][64] -> bf16 [64][K]  (K = 576 or 1152)
// ---------------------------------------------------------------------------
__global__ void transpose_w_kernel(const float* __restrict__ w,
                                   bf16* __restrict__ wt, int KK) {
    int idx = blockIdx.x * 256 + threadIdx.x;
    if (idx < KK * 64) {
        int co = idx / KK;
        int k  = idx - co * KK;
        wt[idx] = (bf16)w[k * 64 + co];
    }
}

// ---------------------------------------------------------------------------
// Kernel 1: conv3x3 SAME, NHWC f32 in (Cin=64), wt = bf16 [64][576].
// Writes BOTH bf16 layouts: out  = [B,C,T,F] (d-contiguous rows)
//                           outT = [B,C,F,T] (t-contiguous rows)
// ---------------------------------------------------------------------------
__global__ __launch_bounds__(128) void conv_in_kernel(
    const float* __restrict__ x, const bf16* __restrict__ wt,
    const float* __restrict__ bias, bf16* __restrict__ out,
    bf16* __restrict__ outT) {
    __shared__ alignas(16) bf16 sx[3 * 18 * 64];  // patch t-1..t+1, f0-1..f0+16

    int m0  = blockIdx.x * 16;
    int b   = m0 / TF_;
    int rem = m0 - b * TF_;
    int t   = rem / F_;
    int f0  = rem - t * F_;  // multiple of 16

    // stage input patch as bf16 (float4 loads, packed bf16 stores)
    for (int e = threadIdx.x; e < 3 * 18 * 16; e += 128) {
        int ky = e / (18 * 16);
        int r2 = e - ky * 18 * 16;
        int fx = r2 >> 4;
        int c4 = (r2 & 15) * 4;
        int tt = t + ky - 1;
        int ff = f0 + fx - 1;
        union { bf16 bb[4]; uint2 u; } pk;
        if ((unsigned)tt < (unsigned)T_ && (unsigned)ff < (unsigned)F_) {
            float4 v = *(const float4*)(x + (((size_t)b * T_ + tt) * F_ + ff) * 64 + c4);
            pk.bb[0] = (bf16)v.x; pk.bb[1] = (bf16)v.y;
            pk.bb[2] = (bf16)v.z; pk.bb[3] = (bf16)v.w;
        } else {
            pk.u.x = 0u; pk.u.y = 0u;
        }
        *(uint2*)(&sx[(ky * 18 + fx) * 64 + c4]) = pk.u;
    }
    __syncthreads();

    int lane = threadIdx.x & 31;
    int wid  = threadIdx.x >> 5;
    int m    = lane & 15;
    int hs   = (lane & 16) ? 8 : 0;
    int co   = wid * 16 + m;

    float bv = bias[co];
    f32x8 acc;
#pragma unroll
    for (int r = 0; r < 8; ++r) acc[r] = bv;

    const bf16* wrow = wt + (size_t)co * 576;
    for (int ks = 0; ks < 18; ++ks) {
        Frag a, bw;
        int K0 = ks * 32 + hs;       // first 8-run
        int K1 = K0 + 16;            // second 8-run
        {
            int tap = K0 >> 6, ci = K0 & 63;
            int ky = tap / 3, kx = tap - ky * 3;
            a.q[0] = *(const uint4*)(&sx[(ky * 18 + m + kx) * 64 + ci]);
        }
        {
            int tap = K1 >> 6, ci = K1 & 63;
            int ky = tap / 3, kx = tap - ky * 3;
            a.q[1] = *(const uint4*)(&sx[(ky * 18 + m + kx) * 64 + ci]);
        }
        bw.q[0] = *(const uint4*)(wrow + K0);
        bw.q[1] = *(const uint4*)(wrow + K1);
        acc = wmma_bf16(a, bw, acc);
    }

    // rows r..r+7 of this lane are 8 consecutive f positions -> packed store
    union { bf16 bb[8]; uint4 u; } po;
#pragma unroll
    for (int r = 0; r < 8; ++r) po.bb[r] = (bf16)acc[r];
    *(uint4*)(out + ((size_t)(b * 64 + co) * T_ + t) * F_ + f0 + hs) = po.u;
#pragma unroll
    for (int r = 0; r < 8; ++r)
        outT[((size_t)(b * 64 + co) * F_ + f0 + hs + r) * T_ + t] = po.bb[r];
}

// ---------------------------------------------------------------------------
// Kernel 2: freq-branch flash attention per (b,c): Q,K = [512,80] rows from
// [B,C,T,F]; V consumed column-major from the [B,C,F,T] copy.  One wave per
// 16-row q-tile, 16 key-steps of 32, online softmax, P relaid through LDS.
// Output scattered into [B,T,F,64].
// ---------------------------------------------------------------------------
__global__ __launch_bounds__(32) void attn_freq_kernel(
    const bf16* __restrict__ cq, const bf16* __restrict__ ck,
    const bf16* __restrict__ cvT, bf16* __restrict__ of) {
    __shared__ alignas(16) bf16 sP[16 * 32];

    int bc   = blockIdx.y;
    int qt   = blockIdx.x;
    int lane = threadIdx.x;
    int m    = lane & 15;
    int hs   = (lane & 16) ? 8 : 0;
    int b    = bc >> 6;
    int c    = bc & 63;

    const bf16* Q  = cq  + (size_t)bc * TF_;
    const bf16* Kp = ck  + (size_t)bc * TF_;
    const bf16* VT = cvT + (size_t)bc * TF_;   // [F][T]
    const float scale = 0.11180339887498949f;  // 1/sqrt(80)
    const uint4 z4 = zero4();

    Frag qa[3];
    const bf16* qp = Q + (size_t)(qt * 16 + m) * F_;
#pragma unroll
    for (int kk = 0; kk < 3; ++kk) {
        qa[kk].q[0] = *(const uint4*)(qp + kk * 32 + hs);
        qa[kk].q[1] = (kk < 2) ? *(const uint4*)(qp + kk * 32 + hs + 16) : z4;
    }

    float mr[8], lr[8];
    f32x8 oacc[5];
#pragma unroll
    for (int r = 0; r < 8; ++r) { mr[r] = -1e30f; lr[r] = 0.f; }
#pragma unroll
    for (int nt = 0; nt < 5; ++nt)
#pragma unroll
        for (int r = 0; r < 8; ++r) oacc[nt][r] = 0.f;

    for (int j = 0; j < 16; ++j) {  // 32 keys per step
        f32x8 s[2];
#pragma unroll
        for (int h = 0; h < 2; ++h) {
#pragma unroll
            for (int r = 0; r < 8; ++r) s[h][r] = 0.f;
            const bf16* kp = Kp + (size_t)(j * 32 + h * 16 + m) * F_;
#pragma unroll
            for (int kk = 0; kk < 3; ++kk) {
                Frag kb;
                kb.q[0] = *(const uint4*)(kp + kk * 32 + hs);
                kb.q[1] = (kk < 2) ? *(const uint4*)(kp + kk * 32 + hs + 16) : z4;
                s[h] = wmma_bf16(qa[kk], kb, s[h]);
            }
#pragma unroll
            for (int r = 0; r < 8; ++r) s[h][r] *= scale;
        }

        float p0[8], p1[8], al[8];
#pragma unroll
        for (int r = 0; r < 8; ++r) {
            float x = fmaxf(s[0][r], s[1][r]);
            x = fmaxf(x, __shfl_xor(x, 1, 32));
            x = fmaxf(x, __shfl_xor(x, 2, 32));
            x = fmaxf(x, __shfl_xor(x, 4, 32));
            x = fmaxf(x, __shfl_xor(x, 8, 32));
            float nm = fmaxf(mr[r], x);
            al[r] = __expf(mr[r] - nm);
            mr[r] = nm;
            p0[r] = __expf(s[0][r] - nm);
            p1[r] = __expf(s[1][r] - nm);
            float su = p0[r] + p1[r];
            su += __shfl_xor(su, 1, 32);
            su += __shfl_xor(su, 2, 32);
            su += __shfl_xor(su, 4, 32);
            su += __shfl_xor(su, 8, 32);
            lr[r] = lr[r] * al[r] + su;
        }
#pragma unroll
        for (int nt = 0; nt < 5; ++nt)
#pragma unroll
            for (int r = 0; r < 8; ++r) oacc[nt][r] *= al[r];

        __syncthreads();  // protect sP from previous iteration's readers
#pragma unroll
        for (int r = 0; r < 8; ++r) {
            int M = r + hs;
            sP[M * 32 + m]      = (bf16)p0[r];
            sP[M * 32 + 16 + m] = (bf16)p1[r];
        }
        __syncthreads();

        Frag pa;  // A-layout read: two contiguous 8-runs -> b128 ds loads
        pa.q[0] = *(const uint4*)(&sP[m * 32 + hs]);
        pa.q[1] = *(const uint4*)(&sP[m * 32 + hs + 16]);

#pragma unroll
        for (int nt = 0; nt < 5; ++nt) {
            Frag vb;  // B[key][d] from [F][T] copy: t-runs contiguous
            const bf16* vtp = VT + (size_t)(nt * 16 + m) * T_ + j * 32;
            vb.q[0] = *(const uint4*)(vtp + hs);
            vb.q[1] = *(const uint4*)(vtp + hs + 16);
            oacc[nt] = wmma_bf16(pa, vb, oacc[nt]);
        }
    }

#pragma unroll
    for (int r = 0; r < 8; ++r) {
        int M = r + hs;
        float inv = 1.f / lr[r];
#pragma unroll
        for (int nt = 0; nt < 5; ++nt)
            of[(((size_t)b * T_ + qt * 16 + M) * F_ + nt * 16 + m) * 64 + c] =
                (bf16)(oacc[nt][r] * inv);
    }
}

// ---------------------------------------------------------------------------
// Kernel 3: time-branch attention per (b,c): Q',K' = [80,512] rows straight
// from the [B,C,F,T] copies (t-contiguous); V' columns from [B,C,T,F].
// 80 keys = 5 S-fragments; full softmax; output scattered into [B,T,F,64].
// ---------------------------------------------------------------------------
__global__ __launch_bounds__(32) void attn_time_kernel(
    const bf16* __restrict__ cqT, const bf16* __restrict__ ckT,
    const bf16* __restrict__ cv, bf16* __restrict__ ot) {
    __shared__ alignas(16) bf16 sP[16 * 96];

    int bc   = blockIdx.y;
    int qt   = blockIdx.x;  // 0..4
    int lane = threadIdx.x;
    int m    = lane & 15;
    int hs   = (lane & 16) ? 8 : 0;
    int b    = bc >> 6;
    int c    = bc & 63;

    const bf16* QT = cqT + (size_t)bc * TF_;   // [F][T]
    const bf16* KT = ckT + (size_t)bc * TF_;   // [F][T]
    const bf16* Vp = cv  + (size_t)bc * TF_;   // [T][F]
    const float scale = 0.044194173824159216f; // 1/sqrt(512)
    const uint4 z4 = zero4();

    f32x8 s[5];
#pragma unroll
    for (int h = 0; h < 5; ++h)
#pragma unroll
        for (int r = 0; r < 8; ++r) s[h][r] = 0.f;

    const bf16* qrow = QT + (size_t)(qt * 16 + m) * T_;
    for (int ks = 0; ks < 16; ++ks) {  // K-dim (t) in steps of 32
        Frag qa;
        qa.q[0] = *(const uint4*)(qrow + ks * 32 + hs);
        qa.q[1] = *(const uint4*)(qrow + ks * 32 + hs + 16);
#pragma unroll
        for (int h = 0; h < 5; ++h) {
            Frag kb;
            const bf16* krow = KT + (size_t)(h * 16 + m) * T_ + ks * 32;
            kb.q[0] = *(const uint4*)(krow + hs);
            kb.q[1] = *(const uint4*)(krow + hs + 16);
            s[h] = wmma_bf16(qa, kb, s[h]);
        }
    }

#pragma unroll
    for (int h = 0; h < 5; ++h)
#pragma unroll
        for (int r = 0; r < 8; ++r) s[h][r] *= scale;

#pragma unroll
    for (int r = 0; r < 8; ++r) {
        float x = s[0][r];
#pragma unroll
        for (int h = 1; h < 5; ++h) x = fmaxf(x, s[h][r]);
        x = fmaxf(x, __shfl_xor(x, 1, 32));
        x = fmaxf(x, __shfl_xor(x, 2, 32));
        x = fmaxf(x, __shfl_xor(x, 4, 32));
        x = fmaxf(x, __shfl_xor(x, 8, 32));
        float su = 0.f;
#pragma unroll
        for (int h = 0; h < 5; ++h) { s[h][r] = __expf(s[h][r] - x); su += s[h][r]; }
        su += __shfl_xor(su, 1, 32);
        su += __shfl_xor(su, 2, 32);
        su += __shfl_xor(su, 4, 32);
        su += __shfl_xor(su, 8, 32);
        float inv = 1.f / su;
#pragma unroll
        for (int h = 0; h < 5; ++h) s[h][r] *= inv;
    }

    // stash normalized P [16 x 96] (cols 80..95 zero) for A-layout reload
    __syncthreads();
#pragma unroll
    for (int r = 0; r < 8; ++r) {
        int M = r + hs;
#pragma unroll
        for (int h = 0; h < 5; ++h) sP[M * 96 + h * 16 + m] = (bf16)s[h][r];
        sP[M * 96 + 80 + m] = (bf16)0.f;
    }
    __syncthreads();

    Frag pa[3];
#pragma unroll
    for (int kk = 0; kk < 3; ++kk) {
        pa[kk].q[0] = *(const uint4*)(&sP[m * 96 + kk * 32 + hs]);
        pa[kk].q[1] = *(const uint4*)(&sP[m * 96 + kk * 32 + hs + 16]);
    }

    for (int nt = 0; nt < 32; ++nt) {  // output t-columns
        f32x8 acc;
#pragma unroll
        for (int r = 0; r < 8; ++r) acc[r] = 0.f;
        const bf16* vp = Vp + (size_t)(nt * 16 + m) * F_;
#pragma unroll
        for (int kk = 0; kk < 3; ++kk) {
            Frag vb;  // B[key_f][t]: f-runs contiguous in [T][F]
            vb.q[0] = *(const uint4*)(vp + kk * 32 + hs);
            vb.q[1] = (kk < 2) ? *(const uint4*)(vp + kk * 32 + hs + 16) : z4;
            acc = wmma_bf16(pa[kk], vb, acc);
        }
#pragma unroll
        for (int r = 0; r < 8; ++r) {
            int M = r + hs;
            ot[(((size_t)b * T_ + nt * 16 + m) * F_ + qt * 16 + M) * 64 + c] =
                (bf16)acc[r];
        }
    }
}

// ---------------------------------------------------------------------------
// Kernel 4: conv3x3 SAME over concat([time, freq]) channels (128 -> 64),
// inputs bf16 [B,T,F,64] x2, wt = bf16 [64][1152], f32 NHWC output.
// ---------------------------------------------------------------------------
__global__ __launch_bounds__(128) void conv_out_kernel(
    const bf16* __restrict__ att_t, const bf16* __restrict__ att_f,
    const bf16* __restrict__ wt, const float* __restrict__ bias,
    float* __restrict__ out) {
    __shared__ alignas(16) bf16 sx[3 * 18 * 128];

    int m0  = blockIdx.x * 16;
    int b   = m0 / TF_;
    int rem = m0 - b * TF_;
    int t   = rem / F_;
    int f0  = rem - t * F_;

    for (int e = threadIdx.x; e < 3 * 18 * 16; e += 128) {
        int ky = e / (18 * 16);
        int r2 = e - ky * 18 * 16;
        int fx = r2 >> 4;
        int c8 = (r2 & 15) * 8;
        int tt = t + ky - 1;
        int ff = f0 + fx - 1;
        uint4 val = zero4();
        if ((unsigned)tt < (unsigned)T_ && (unsigned)ff < (unsigned)F_) {
            const bf16* src = (c8 < 64) ? att_t : att_f;
            int ci = c8 & 63;
            val = *(const uint4*)(src + (((size_t)b * T_ + tt) * F_ + ff) * 64 + ci);
        }
        *(uint4*)(&sx[(ky * 18 + fx) * 128 + c8]) = val;
    }
    __syncthreads();

    int lane = threadIdx.x & 31;
    int wid  = threadIdx.x >> 5;
    int m    = lane & 15;
    int hs   = (lane & 16) ? 8 : 0;
    int co   = wid * 16 + m;

    float bv = bias[co];
    f32x8 acc;
#pragma unroll
    for (int r = 0; r < 8; ++r) acc[r] = bv;

    const bf16* wrow = wt + (size_t)co * 1152;
    for (int ks = 0; ks < 36; ++ks) {
        Frag a, bw;
        int K0 = ks * 32 + hs;
        int K1 = K0 + 16;
        {
            int tap = K0 >> 7, cc = K0 & 127;
            int ky = tap / 3, kx = tap - ky * 3;
            a.q[0] = *(const uint4*)(&sx[(ky * 18 + m + kx) * 128 + cc]);
        }
        {
            int tap = K1 >> 7, cc = K1 & 127;
            int ky = tap / 3, kx = tap - ky * 3;
            a.q[1] = *(const uint4*)(&sx[(ky * 18 + m + kx) * 128 + cc]);
        }
        bw.q[0] = *(const uint4*)(wrow + K0);
        bw.q[1] = *(const uint4*)(wrow + K1);
        acc = wmma_bf16(a, bw, acc);
    }

#pragma unroll
    for (int r = 0; r < 8; ++r) {
        int M = r + hs;
        out[(((size_t)b * T_ + t) * F_ + (f0 + M)) * 64 + co] = acc[r];
    }
}

// ---------------------------------------------------------------------------
extern "C" void kernel_launch(void* const* d_in, const int* in_sizes, int n_in,
                              void* d_out, int out_size, void* d_ws, size_t ws_size,
                              hipStream_t stream) {
    (void)in_sizes; (void)n_in; (void)out_size; (void)ws_size;

    const float* q  = (const float*)d_in[0];
    const float* k  = (const float*)d_in[1];
    const float* v  = (const float*)d_in[2];
    const float* wq = (const float*)d_in[3];
    const float* bq = (const float*)d_in[4];
    const float* wk = (const float*)d_in[5];
    const float* bk = (const float*)d_in[6];
    const float* wv = (const float*)d_in[7];
    const float* bv = (const float*)d_in[8];
    const float* wo = (const float*)d_in[9];
    const float* bo = (const float*)d_in[10];

    const size_t SZ = (size_t)B_ * 64 * T_ * F_;  // 20,971,520 bf16 elems per buffer
    bf16* ws    = (bf16*)d_ws;
    bf16* cq    = ws;                 // [B,C,T,F]
    bf16* ck    = ws + SZ;            // [B,C,T,F]
    bf16* cv    = ws + 2 * SZ;        // [B,C,T,F]
    bf16* cqT   = ws + 3 * SZ;        // [B,C,F,T]
    bf16* ckT   = ws + 4 * SZ;        // [B,C,F,T]
    bf16* cvT   = ws + 5 * SZ;        // [B,C,F,T]
    bf16* att_t = ws + 6 * SZ;        // [B,T,F,64]
    bf16* att_f = ws + 7 * SZ;        // [B,T,F,64]
    bf16* wtq   = ws + 8 * SZ;        // [64][576]
    bf16* wtk   = wtq + 576 * 64;
    bf16* wtv   = wtk + 576 * 64;
    bf16* wto   = wtv + 576 * 64;     // [64][1152]

    transpose_w_kernel<<<(576 * 64 + 255) / 256, 256, 0, stream>>>(wq, wtq, 576);
    transpose_w_kernel<<<(576 * 64 + 255) / 256, 256, 0, stream>>>(wk, wtk, 576);
    transpose_w_kernel<<<(576 * 64 + 255) / 256, 256, 0, stream>>>(wv, wtv, 576);
    transpose_w_kernel<<<(1152 * 64 + 255) / 256, 256, 0, stream>>>(wo, wto, 1152);

    dim3 gconv(B_ * T_ * F_ / 16);  // 20480 M-tiles
    dim3 bconv(128);                 // 4 waves, one 16-wide N-tile each

    conv_in_kernel<<<gconv, bconv, 0, stream>>>(q, wtq, bq, cq, cqT);
    conv_in_kernel<<<gconv, bconv, 0, stream>>>(k, wtk, bk, ck, ckT);
    conv_in_kernel<<<gconv, bconv, 0, stream>>>(v, wtv, bv, cv, cvT);

    attn_freq_kernel<<<dim3(T_ / 16, B_ * 64), 32, 0, stream>>>(cq, ck, cvT, att_f);
    attn_time_kernel<<<dim3(F_ / 16, B_ * 64), 32, 0, stream>>>(cqT, ckT, cv, att_t);

    conv_out_kernel<<<gconv, bconv, 0, stream>>>(att_t, att_f, wto, bo, (float*)d_out);
}
